// Attention_22290880266784
// MI455X (gfx1250) — matrix operand order repaired and checked by reading, not compiled, not run on hardware
//
#include <hip/hip_runtime.h>
#include <hip/hip_bf16.h>
#include <cstdint>
#include <cstddef>

// ---------------------------------------------------------------------------
// Types for CDNA5 WMMA (wave32, 16x16x32 bf16 -> f32)
// ---------------------------------------------------------------------------
typedef __bf16 bf16_t;
typedef __attribute__((ext_vector_type(16))) __bf16 bf16x16;
typedef __attribute__((ext_vector_type(8)))  __bf16 bf16x8;
typedef __attribute__((ext_vector_type(4)))  __bf16 bf16x4;
typedef __attribute__((ext_vector_type(8)))  float  f32x8;
typedef __attribute__((ext_vector_type(4)))  float  f32x4;
typedef int i32x4v __attribute__((vector_size(4 * sizeof(int))));  // matches builtin

constexpr int BB = 4;       // batch
constexpr int SS = 2048;    // sequence
constexpr int DD = 1024;    // model dim
constexpr int BM = 128;     // block tile M (8 waves x 16 rows)
constexpr int BN = 64;      // block tile N (4 accumulators x 16 cols)
constexpr int BK = 64;      // k-step per stage (2 x WMMA K)
constexpr int LP = 72;      // LDS row pitch in bf16 (64 + 8 pad, 144B rows)

// ---------------------------------------------------------------------------
// gfx1250 async global->LDS copy path (ASYNCcnt), with synchronous fallback
// ---------------------------------------------------------------------------
#define AS1 __attribute__((address_space(1)))
#define AS3 __attribute__((address_space(3)))

#if defined(__AMDGCN__) && __has_builtin(__builtin_amdgcn_global_load_async_to_lds_b128)
#define HAS_ASYNC_LDS 1
#else
#define HAS_ASYNC_LDS 0
#endif

#if HAS_ASYNC_LDS
#define ASYNC_COPY_B128(gp, lp)                                              \
  __builtin_amdgcn_global_load_async_to_lds_b128(                            \
      (AS1 i32x4v*)(uintptr_t)(gp), (AS3 i32x4v*)(uint32_t)(uintptr_t)(lp), 0, 0)
#if __has_builtin(__builtin_amdgcn_s_wait_asynccnt)
#define STAGE_FENCE() __builtin_amdgcn_s_wait_asynccnt(0)
#else
#define STAGE_FENCE() asm volatile("s_wait_asynccnt 0x0" ::: "memory")
#endif
#else
#define ASYNC_COPY_B128(gp, lp) (*(bf16x8*)(lp) = *(const bf16x8*)(gp))
#define STAGE_FENCE() ((void)0)
#endif

// ---------------------------------------------------------------------------
// Staging: global -> LDS tiles of [ROWS][BK], 256 threads, fully unrolled
// ---------------------------------------------------------------------------
template <int ROWS>
__device__ __forceinline__ void stage_f32_bf16(const float* __restrict__ g,
                                               size_t gpitch, bf16_t* lds, int tid) {
  constexpr int NCH = ROWS * (BK / 4);  // chunks of 4 floats
#pragma unroll
  for (int c = tid; c < NCH; c += 256) {
    int row = c >> 4, cg = c & 15;      // BK/4 == 16 chunks per row
    f32x4 v = *(const f32x4*)(g + (size_t)row * gpitch + cg * 4);
    bf16x4 o;
    o[0] = (bf16_t)v[0]; o[1] = (bf16_t)v[1];
    o[2] = (bf16_t)v[2]; o[3] = (bf16_t)v[3];
    *(bf16x4*)(lds + row * LP + cg * 4) = o;
  }
}

template <int ROWS>
__device__ __forceinline__ void stage_bf16(const bf16_t* __restrict__ g,
                                           size_t gpitch, bf16_t* lds, int tid) {
  constexpr int NCH = ROWS * (BK / 8);  // chunks of 8 bf16 (16B)
#pragma unroll
  for (int c = tid; c < NCH; c += 256) {
    int row = c >> 3, cg = c & 7;       // BK/8 == 8 chunks per row
    ASYNC_COPY_B128(g + (size_t)row * gpitch + cg * 8, lds + row * LP + cg * 8);
  }
}

// ---------------------------------------------------------------------------
// WMMA fragments (ISA 7.12.2 layouts, wave32); h selects 32-wide k-half
// A 16x32: lanes 0-15 row=lane, K={0..7,16..23}; lanes 16-31 row=lane-16, K={8..15,24..31}
// B 32x16: lanes 0-15 col=lane, K={0..7,16..23}; lanes 16-31 col=lane-16, K={8..15,24..31}
// ---------------------------------------------------------------------------
__device__ __forceinline__ bf16x16 frag_a(const bf16_t* lds, int row, int lane, int h) {
  int base = row * LP + h * 32 + ((lane & 16) ? 8 : 0);
  bf16x8 a0 = *(const bf16x8*)(lds + base);
  bf16x8 a1 = *(const bf16x8*)(lds + base + 16);
  bf16x16 r;
#pragma unroll
  for (int i = 0; i < 8; ++i) { r[i] = a0[i]; r[i + 8] = a1[i]; }
  return r;
}

__device__ __forceinline__ bf16x16 frag_b(const bf16_t* lds, int col, int lane, int h) {
  int base = col * LP + h * 32 + ((lane & 16) ? 8 : 0);
  bf16x8 b0 = *(const bf16x8*)(lds + base);
  bf16x8 b1 = *(const bf16x8*)(lds + base + 16);
  bf16x16 r;
#pragma unroll
  for (int i = 0; i < 8; ++i) { r[i] = b0[i]; r[i + 8] = b1[i]; }
  return r;
}

__device__ __forceinline__ f32x8 wmma_bf16(bf16x16 a, bf16x16 b, f32x8 c) {
  // (neg_a, A, neg_b, B, c_mod, C, reuse_a, reuse_b)
  return __builtin_amdgcn_wmma_f32_16x16x32_bf16(false, a, false, b, (short)0, c,
                                                 false, false);
}

// Load ALL fragments first (20 x ds_load_b128), then run the 8-WMMA chain so
// the scheduler can use partial dscnt waits and keep the matrix pipe busy.
__device__ __forceinline__ void compute_tile(const bf16_t* Abuf, const bf16_t* Bbuf,
                                             int wrow, int lane, f32x8 (&acc)[4]) {
  bf16x16 af[2];
  bf16x16 bfv[2][4];
#pragma unroll
  for (int h = 0; h < 2; ++h) {
    af[h] = frag_a(Abuf, wrow + (lane & 15), lane, h);
#pragma unroll
    for (int j = 0; j < 4; ++j)
      bfv[h][j] = frag_b(Bbuf, j * 16 + (lane & 15), lane, h);
  }
#pragma unroll
  for (int h = 0; h < 2; ++h)
#pragma unroll
    for (int j = 0; j < 4; ++j)
      acc[j] = wmma_bf16(af[h], bfv[h][j], acc[j]);
}

// ---------------------------------------------------------------------------
// Kernel 1: projection  C[b,s,e] = sum_d X[b,s,d]*W[e,d] + bias[e]
// transpose_out==0 : out bf16 [b][s][e] (pitch DD)   (Q, K)
// transpose_out==1 : out bf16 [b][e][s] (pitch SS)   (V transposed)
// Double-buffered; converts fp32->bf16 during staging.
// ---------------------------------------------------------------------------
__global__ __launch_bounds__(256) void proj_kernel(const float* __restrict__ X,
                                                   const float* __restrict__ W,
                                                   const float* __restrict__ bias,
                                                   bf16_t* __restrict__ out,
                                                   int transpose_out) {
  __shared__ bf16_t As[2][BM * LP];
  __shared__ bf16_t Bs[2][BN * LP];
  const int tid  = threadIdx.x;
  const int m0   = blockIdx.x * BM;
  const int n0   = blockIdx.y * BN;
  const int b    = blockIdx.z;
  const int wave = tid >> 5, lane = tid & 31;
  const int wrow = wave * 16;

  const float* Xb = X + ((size_t)b * SS + m0) * DD;
  const float* Wb = W + (size_t)n0 * DD;

  f32x8 acc[4];
#pragma unroll
  for (int j = 0; j < 4; ++j)
#pragma unroll
    for (int i = 0; i < 8; ++i) acc[j][i] = 0.0f;

  constexpr int NK = DD / BK;
  stage_f32_bf16<BM>(Xb, DD, As[0], tid);
  stage_f32_bf16<BN>(Wb, DD, Bs[0], tid);
  __syncthreads();
  for (int kt = 0; kt < NK; ++kt) {
    const int cur = kt & 1;
    if (kt + 1 < NK) {
      stage_f32_bf16<BM>(Xb + (kt + 1) * BK, DD, As[cur ^ 1], tid);
      stage_f32_bf16<BN>(Wb + (kt + 1) * BK, DD, Bs[cur ^ 1], tid);
    }
    compute_tile(As[cur], Bs[cur], wrow, lane, acc);
    __syncthreads();
  }

  const int rbase = m0 + wrow + ((lane & 16) ? 8 : 0);
#pragma unroll
  for (int j = 0; j < 4; ++j) {
    const int col = n0 + j * 16 + (lane & 15);
    const float bv = bias[col];
    if (!transpose_out) {
#pragma unroll
      for (int v = 0; v < 8; ++v)
        out[((size_t)b * SS + (rbase + v)) * DD + col] = (bf16_t)(acc[j][v] + bv);
    } else {
      bf16x8 pk;
#pragma unroll
      for (int v = 0; v < 8; ++v) pk[v] = (bf16_t)(acc[j][v] + bv);
      *(bf16x8*)(out + ((size_t)b * DD + col) * SS + rbase) = pk;
    }
  }
}

// ---------------------------------------------------------------------------
// Kernel 2: scores  Sc[b,q,j] = (1/32) * sum_d Q[b,q,d]*K[b,j,d]   (fp32 out)
// Async double-buffered staging; blocks above the causal diagonal exit early.
// ---------------------------------------------------------------------------
__global__ __launch_bounds__(256) void scores_kernel(const bf16_t* __restrict__ Q,
                                                     const bf16_t* __restrict__ K,
                                                     float* __restrict__ Sc) {
  const int q0 = blockIdx.x * BM;
  const int j0 = blockIdx.y * BN;
  if (j0 > q0 + (BM - 1)) return;  // entirely above diagonal
  const int b = blockIdx.z;

  __shared__ bf16_t As[2][BM * LP];
  __shared__ bf16_t Bs[2][BN * LP];
  const int tid = threadIdx.x;
  const int wave = tid >> 5, lane = tid & 31;
  const int wrow = wave * 16;

  const bf16_t* Qb = Q + ((size_t)b * SS + q0) * DD;
  const bf16_t* Kb = K + ((size_t)b * SS + j0) * DD;

  f32x8 acc[4];
#pragma unroll
  for (int j = 0; j < 4; ++j)
#pragma unroll
    for (int i = 0; i < 8; ++i) acc[j][i] = 0.0f;

  constexpr int NK = DD / BK;
  stage_bf16<BM>(Qb, DD, As[0], tid);
  stage_bf16<BN>(Kb, DD, Bs[0], tid);
  STAGE_FENCE();
  __syncthreads();
  for (int kt = 0; kt < NK; ++kt) {
    const int cur = kt & 1;
    if (kt + 1 < NK) {
      stage_bf16<BM>(Qb + (kt + 1) * BK, DD, As[cur ^ 1], tid);
      stage_bf16<BN>(Kb + (kt + 1) * BK, DD, Bs[cur ^ 1], tid);
    }
    compute_tile(As[cur], Bs[cur], wrow, lane, acc);
    STAGE_FENCE();
    __syncthreads();
  }

  const float scale = 0.03125f;  // 1/sqrt(1024)
  const int rbase = q0 + wrow + ((lane & 16) ? 8 : 0);
#pragma unroll
  for (int j = 0; j < 4; ++j) {
    const int col = j0 + j * 16 + (lane & 15);
#pragma unroll
    for (int v = 0; v < 8; ++v)
      Sc[((size_t)b * SS + (rbase + v)) * SS + col] = acc[j][v] * scale;
  }
}

// ---------------------------------------------------------------------------
// Kernel 3: causal softmax over row q (length q+1); writes bf16 probabilities
// in-place into the same rows (byte pitch unchanged: SS*4 bytes = 2*SS bf16).
// Columns j > q get exact zeros. Never reads skipped score tiles.
// ---------------------------------------------------------------------------
__global__ __launch_bounds__(256) void softmax_kernel(float* __restrict__ Sc) {
  const int r = blockIdx.x;        // r in [0, BB*SS)
  const int q = r & (SS - 1);
  float* row = Sc + (size_t)r * SS;
  bf16_t* prow = (bf16_t*)row;
  const int tid = threadIdx.x;

  float x[8];
#pragma unroll
  for (int i = 0; i < 8; ++i) {
    int j = tid + i * 256;
    x[i] = (j <= q) ? row[j] : -3.0e38f;
  }

  __shared__ float red[256];
  float m = x[0];
#pragma unroll
  for (int i = 1; i < 8; ++i) m = fmaxf(m, x[i]);
  red[tid] = m;
  __syncthreads();
  for (int s = 128; s > 0; s >>= 1) {
    if (tid < s) red[tid] = fmaxf(red[tid], red[tid + s]);
    __syncthreads();
  }
  m = red[0];
  __syncthreads();

  float e[8], sum = 0.0f;
#pragma unroll
  for (int i = 0; i < 8; ++i) {
    int j = tid + i * 256;
    e[i] = (j <= q) ? __expf(x[i] - m) : 0.0f;
    sum += e[i];
  }
  red[tid] = sum;
  __syncthreads();
  for (int s = 128; s > 0; s >>= 1) {
    if (tid < s) red[tid] += red[tid + s];
    __syncthreads();
  }
  const float inv = 1.0f / red[0];
  __syncthreads();  // all reads of `row` completed before bf16 overwrite

#pragma unroll
  for (int i = 0; i < 8; ++i) prow[tid + i * 256] = (bf16_t)(e[i] * inv);
}

// ---------------------------------------------------------------------------
// Kernel 4: output  O[b,q,e] = sum_j P[b,q,j] * Vt[b,e,j]   (fp32 out)
// P rows have bf16 element pitch 2*SS (aliased over the fp32 score buffer).
// j-loop clipped at the causal boundary of the q-tile (P is zero beyond q).
// ---------------------------------------------------------------------------
__global__ __launch_bounds__(256) void out_kernel(const bf16_t* __restrict__ P,
                                                  const bf16_t* __restrict__ Vt,
                                                  float* __restrict__ out) {
  const int q0 = blockIdx.x * BM;
  const int e0 = blockIdx.y * BN;
  const int b  = blockIdx.z;

  __shared__ bf16_t As[2][BM * LP];
  __shared__ bf16_t Bs[2][BN * LP];
  const int tid = threadIdx.x;
  const int wave = tid >> 5, lane = tid & 31;
  const int wrow = wave * 16;

  const size_t ppitch = (size_t)2 * SS;  // bf16 elements per score row
  const bf16_t* Pb = P + ((size_t)b * SS + q0) * ppitch;
  const bf16_t* Vb = Vt + ((size_t)b * DD + e0) * SS;

  f32x8 acc[4];
#pragma unroll
  for (int j = 0; j < 4; ++j)
#pragma unroll
    for (int i = 0; i < 8; ++i) acc[j][i] = 0.0f;

  const int nk = (q0 + BM) / BK;  // causal: P[q][j]==0 for j >= q0+BM
  stage_bf16<BM>(Pb, ppitch, As[0], tid);
  stage_bf16<BN>(Vb, SS, Bs[0], tid);
  STAGE_FENCE();
  __syncthreads();
  for (int kt = 0; kt < nk; ++kt) {
    const int cur = kt & 1;
    if (kt + 1 < nk) {
      stage_bf16<BM>(Pb + (size_t)(kt + 1) * BK, ppitch, As[cur ^ 1], tid);
      stage_bf16<BN>(Vb + (size_t)(kt + 1) * BK, SS, Bs[cur ^ 1], tid);
    }
    compute_tile(As[cur], Bs[cur], wrow, lane, acc);
    STAGE_FENCE();
    __syncthreads();
  }

  const int rbase = q0 + wrow + ((lane & 16) ? 8 : 0);
#pragma unroll
  for (int j = 0; j < 4; ++j) {
    const int col = e0 + j * 16 + (lane & 15);
#pragma unroll
    for (int v = 0; v < 8; ++v)
      out[((size_t)b * SS + (rbase + v)) * DD + col] = acc[j][v];
  }
}

// ---------------------------------------------------------------------------
// Host-side launcher
// inputs: q, k, v, mask, w_q, b_q, w_k, b_k, w_v, b_v
// ws layout: Qb(16MB) | Kb(16MB) | Vt(16MB) | Sc(64MB, aliased as bf16 P)
// ---------------------------------------------------------------------------
extern "C" void kernel_launch(void* const* d_in, const int* in_sizes, int n_in,
                              void* d_out, int out_size, void* d_ws, size_t ws_size,
                              hipStream_t stream) {
  const float* q  = (const float*)d_in[0];
  const float* k  = (const float*)d_in[1];
  const float* v  = (const float*)d_in[2];
  // d_in[3] = additive causal mask; applied analytically in softmax_kernel
  const float* wq = (const float*)d_in[4];
  const float* bq = (const float*)d_in[5];
  const float* wk = (const float*)d_in[6];
  const float* bk = (const float*)d_in[7];
  const float* wv = (const float*)d_in[8];
  const float* bv = (const float*)d_in[9];

  char* ws = (char*)d_ws;
  const size_t qkSz = (size_t)BB * SS * DD * sizeof(bf16_t);  // 16 MB
  bf16_t* Qb = (bf16_t*)(ws);
  bf16_t* Kb = (bf16_t*)(ws + qkSz);
  bf16_t* Vt = (bf16_t*)(ws + 2 * qkSz);
  float*  Sc = (float*)(ws + 3 * qkSz);

  dim3 blk(256);
  dim3 gproj(SS / BM, DD / BN, BB);
  proj_kernel<<<gproj, blk, 0, stream>>>(q, wq, bq, Qb, 0);
  proj_kernel<<<gproj, blk, 0, stream>>>(k, wk, bk, Kb, 0);
  proj_kernel<<<gproj, blk, 0, stream>>>(v, wv, bv, Vt, 1);

  dim3 gsc(SS / BM, SS / BN, BB);
  scores_kernel<<<gsc, blk, 0, stream>>>(Qb, Kb, Sc);

  softmax_kernel<<<BB * SS, blk, 0, stream>>>(Sc);

  dim3 gout(SS / BM, DD / BN, BB);
  out_kernel<<<gout, blk, 0, stream>>>((const bf16_t*)Sc, Vt, (float*)d_out);
}
